// DeepWDK_23974507446892
// MI455X (gfx1250) — matrix operand order repaired and checked by reading, not compiled, not run on hardware
//
#include <hip/hip_runtime.h>
#include <hip/hip_bf16.h>

typedef __attribute__((ext_vector_type(16))) _Float16 v16h;
typedef __attribute__((ext_vector_type(8)))  float    v8f;

#define N_AA 20
#define DDIM 64
#define E_DIM 32
#define LSEQ 200
#define NS 512                 // N1 == N2 == 512
#define KDIM (LSEQ * E_DIM)    // 6400
#define ND   (N_AA * DDIM)     // 1280
#define KBIG (2 * LSEQ * N_AA) // 8000

// ---------------------------------------------------------------------------
// Kernel 0: transpose + convert W (f32, KDIM x ND row-major) -> Wt (f16, ND x KDIM)
// so the WMMA B-fragment (per-lane contiguous K at fixed column n) becomes
// 128-bit global loads.
// ---------------------------------------------------------------------------
__global__ __launch_bounds__(256) void transpose_w(const float* __restrict__ W,
                                                   _Float16* __restrict__ Wt) {
  __shared__ float tileLds[32][33];
  const int tilesN = ND / 32;                 // 40
  int bx = blockIdx.x % tilesN;               // tile along n
  int by = blockIdx.x / tilesN;               // tile along k (200 tiles)
  int tx = threadIdx.x & 31;
  int ty = threadIdx.x >> 5;                  // 0..7
#pragma unroll
  for (int r = 0; r < 4; ++r) {
    int k = by * 32 + ty + 8 * r;
    tileLds[ty + 8 * r][tx] = W[(size_t)k * ND + bx * 32 + tx];
  }
  __syncthreads();
#pragma unroll
  for (int r = 0; r < 4; ++r) {
    int n = bx * 32 + ty + 8 * r;
    Wt[(size_t)n * KDIM + by * 32 + tx] = (_Float16)tileLds[tx][ty + 8 * r];
  }
}

// ---------------------------------------------------------------------------
// Kernel 1: V = gather(E, X) @ W via WMMA f16.  M=512, K=6400, N=1280.
// K-step 32 == E_DIM: each K block maps to exactly one sequence position l, so
// A[m, kBase+k] = E[X[m,l], k] with one index load per lane per step.
// One wave computes a 16x64 strip: the gathered A fragment is reused across
// 4 B fragments -> 4 independent WMMAs per step (no accumulator RAW chains).
// Tiles: (512/16) x (1280/64) = 32 x 20 = 640 waves = 80 blocks.
// ---------------------------------------------------------------------------
__global__ __launch_bounds__(256) void net_gemm(const int* __restrict__ X,
                                                const float* __restrict__ E,
                                                const _Float16* __restrict__ Wt,
                                                float* __restrict__ V) {
  __shared__ _Float16 eLds[N_AA * E_DIM];
  for (int t = threadIdx.x; t < N_AA * E_DIM; t += blockDim.x)
    eLds[t] = (_Float16)E[t];
  __syncthreads();

  const int lane = threadIdx.x & 31;
  const int wave = threadIdx.x >> 5;
  const int tilesN = ND / 64;                 // 20
  const int tile = blockIdx.x * 8 + wave;     // 640 waves total
  const int tm = tile / tilesN;
  const int tn = tile % tilesN;

  const int mrow   = tm * 16 + (lane & 15);   // A row (lanes 0-15 & 16-31: M=lane%16)
  const int ncol   = tn * 64 + (lane & 15);   // first B column for this lane
  const int khalfA = (lane < 16) ? 0 : 16;    // A K-group select
  const int khalfB = (lane < 16) ? 0 : 16;    // B K-half select

  const int*      xrow = X + (size_t)mrow * LSEQ;
  const _Float16* wbase = Wt + (size_t)ncol * KDIM + khalfB;

  v8f acc0 = {}, acc1 = {}, acc2 = {}, acc3 = {};
  for (int l = 0; l < LSEQ; ++l) {
    const int c = xrow[l];
    const _Float16* ev = eLds + c * E_DIM;
    v16h afrag;
#pragma unroll
    for (int e = 0; e < 16; ++e) {
      // A order used here (contiguous, same as B): element e -> K = e + (lane<16?0:16)
      int kk = e + khalfA;
      afrag[e] = ev[kk];
    }
    v16h b0, b1, b2, b3;
    const _Float16* wp = wbase + (size_t)l * E_DIM;
    *((uint4*)&b0)       = *(const uint4*)(wp);
    *(((uint4*)&b0) + 1) = *(const uint4*)(wp + 8);
    *((uint4*)&b1)       = *(const uint4*)(wp + 16 * KDIM);
    *(((uint4*)&b1) + 1) = *(const uint4*)(wp + 16 * KDIM + 8);
    *((uint4*)&b2)       = *(const uint4*)(wp + 32 * KDIM);
    *(((uint4*)&b2) + 1) = *(const uint4*)(wp + 32 * KDIM + 8);
    *((uint4*)&b3)       = *(const uint4*)(wp + 48 * KDIM);
    *(((uint4*)&b3) + 1) = *(const uint4*)(wp + 48 * KDIM + 8);
    acc0 = __builtin_amdgcn_wmma_f32_16x16x32_f16(false, afrag, false, b0,
                                                  (short)0, acc0, false, false);
    acc1 = __builtin_amdgcn_wmma_f32_16x16x32_f16(false, afrag, false, b1,
                                                  (short)0, acc1, false, false);
    acc2 = __builtin_amdgcn_wmma_f32_16x16x32_f16(false, afrag, false, b2,
                                                  (short)0, acc2, false, false);
    acc3 = __builtin_amdgcn_wmma_f32_16x16x32_f16(false, afrag, false, b3,
                                                  (short)0, acc3, false, false);
  }
  // C layout: VGPR r -> row (r + (lane<16?0:8)), col = lane%16
  const int rowOff = (lane < 16) ? 0 : 8;
#pragma unroll
  for (int r = 0; r < 8; ++r) {
    size_t row = (size_t)(tm * 16 + rowOff + r) * ND + tn * 64 + (lane & 15);
    V[row]      = acc0[r];
    V[row + 16] = acc1[r];
    V[row + 32] = acc2[r];
    V[row + 48] = acc3[r];
  }
}

// ---------------------------------------------------------------------------
// Kernel 2: per-sample Gram S[i] = V[i] (20x64) @ V[i]^T  -> (512, 20, 20) f32.
// Tiny (52 MFLOP total): one block per sample, V row cached in LDS.
// ---------------------------------------------------------------------------
__global__ __launch_bounds__(256) void gram(const float* __restrict__ V,
                                            float* __restrict__ S) {
  const int i = blockIdx.x;
  __shared__ float vLds[ND];
  const float* vp = V + (size_t)i * ND;
  for (int t = threadIdx.x; t < ND; t += blockDim.x) vLds[t] = vp[t];
  __syncthreads();
  for (int o = threadIdx.x; o < N_AA * N_AA; o += blockDim.x) {
    int a = o / N_AA, b = o % N_AA;
    float s = 0.f;
#pragma unroll 8
    for (int d = 0; d < DDIM; ++d) s += vLds[a * DDIM + d] * vLds[b * DDIM + d];
    S[(size_t)i * (N_AA * N_AA) + o] = s;
  }
}

// ---------------------------------------------------------------------------
// Kernel 3: diagonal sums k[i] = sum_l S[i, X[i,l], X[i,l]]
// ---------------------------------------------------------------------------
__global__ void kdiag(const int* __restrict__ X, const float* __restrict__ S,
                      float* __restrict__ kout, int n) {
  int i = blockIdx.x * blockDim.x + threadIdx.x;
  if (i >= n) return;
  float s = 0.f;
  for (int l = 0; l < LSEQ; ++l) {
    int c = X[(size_t)i * LSEQ + l];
    s += S[(size_t)i * (N_AA * N_AA) + c * (N_AA + 1)];
  }
  kout[i] = s;
}

// ---------------------------------------------------------------------------
// Kernels 4/5: build f16 GEMM operands for the cross-kernel.
//   Abig[i] = [ P1[i] (4000) | Q2[i] (4000) ]
//     P1[i, l*20+b] = S1[i, X1[i,l], b],   Q2[i, l*20+a] = 1[X1[i,l] == a]
//   Bbig[j] = [ Q1[j] (4000) | P2[j] (4000) ]
//     Q1[j, l*20+b] = 1[X2[j,l] == b],     P2[j, l*20+a] = S2[j, a, X2[j,l]]
// Then K_raw = Abig @ Bbig^T reproduces sum_l (S1[..] + S2[..]).
// ---------------------------------------------------------------------------
__global__ void build_A(const int* __restrict__ X1, const float* __restrict__ S1,
                        _Float16* __restrict__ Abig) {
  size_t t = (size_t)blockIdx.x * blockDim.x + threadIdx.x;
  if (t >= (size_t)NS * LSEQ * N_AA) return;
  int b = (int)(t % N_AA);
  int l = (int)((t / N_AA) % LSEQ);
  int i = (int)(t / ((size_t)N_AA * LSEQ));
  int c = X1[(size_t)i * LSEQ + l];
  _Float16* row = Abig + (size_t)i * KBIG;
  row[l * N_AA + b] = (_Float16)S1[(size_t)i * (N_AA * N_AA) + c * N_AA + b];
  row[LSEQ * N_AA + l * N_AA + b] = (_Float16)((b == c) ? 1.f : 0.f);
}

__global__ void build_B(const int* __restrict__ X2, const float* __restrict__ S2,
                        _Float16* __restrict__ Bbig) {
  size_t t = (size_t)blockIdx.x * blockDim.x + threadIdx.x;
  if (t >= (size_t)NS * LSEQ * N_AA) return;
  int b = (int)(t % N_AA);
  int l = (int)((t / N_AA) % LSEQ);
  int j = (int)(t / ((size_t)N_AA * LSEQ));
  int c = X2[(size_t)j * LSEQ + l];
  _Float16* row = Bbig + (size_t)j * KBIG;
  row[l * N_AA + b] = (_Float16)((b == c) ? 1.f : 0.f);
  row[LSEQ * N_AA + l * N_AA + b] =
      (_Float16)S2[(size_t)j * (N_AA * N_AA) + b * N_AA + c];   // S2[j, a=b, c]
}

// ---------------------------------------------------------------------------
// Kernel 6: K = 0.5 * a^2 * (Abig @ Bbig^T) / sqrt(k1 k2)  via WMMA f16.
// M = N = 512, K = 8000 (250 steps).  One wave computes a 32x32 tile:
// 2 A-frags x 2 B-frags -> 4 independent WMMAs per step (full pipelining,
// 2x less L2 traffic per FLOP than 16x16 tiling).  256 waves = 32 blocks.
// ---------------------------------------------------------------------------
__global__ __launch_bounds__(256) void cross_gemm(const _Float16* __restrict__ Abig,
                                                  const _Float16* __restrict__ Bbig,
                                                  const float* __restrict__ k1,
                                                  const float* __restrict__ k2,
                                                  const float* __restrict__ a_sc,
                                                  float* __restrict__ Kout) {
  const int lane = threadIdx.x & 31;
  const int wave = threadIdx.x >> 5;
  const int tilesN = NS / 32;                 // 16
  const int tile = blockIdx.x * 8 + wave;     // 256 tiles total
  const int tm = tile / tilesN;
  const int tn = tile % tilesN;

  const int irow = tm * 32 + (lane & 15);
  const int jrow = tn * 32 + (lane & 15);
  const int offA = (lane < 16) ? 0 : 8;
  const int offB = (lane < 16) ? 0 : 8;
  const _Float16* ap0 = Abig + (size_t)irow * KBIG;
  const _Float16* ap1 = ap0 + (size_t)16 * KBIG;
  const _Float16* bp0 = Bbig + (size_t)jrow * KBIG;
  const _Float16* bp1 = bp0 + (size_t)16 * KBIG;

  v8f acc00 = {}, acc01 = {}, acc10 = {}, acc11 = {};
  for (int kb = 0; kb < KBIG; kb += 32) {
    v16h a0, a1, b0, b1;
    *((uint4*)&a0)       = *(const uint4*)(ap0 + kb + offA);
    *(((uint4*)&a0) + 1) = *(const uint4*)(ap0 + kb + 16 + offA);
    *((uint4*)&a1)       = *(const uint4*)(ap1 + kb + offA);
    *(((uint4*)&a1) + 1) = *(const uint4*)(ap1 + kb + 16 + offA);
    *((uint4*)&b0)       = *(const uint4*)(bp0 + kb + offB);
    *(((uint4*)&b0) + 1) = *(const uint4*)(bp0 + kb + offB + 16);
    *((uint4*)&b1)       = *(const uint4*)(bp1 + kb + offB);
    *(((uint4*)&b1) + 1) = *(const uint4*)(bp1 + kb + offB + 16);
    acc00 = __builtin_amdgcn_wmma_f32_16x16x32_f16(false, a0, false, b0,
                                                   (short)0, acc00, false, false);
    acc01 = __builtin_amdgcn_wmma_f32_16x16x32_f16(false, a0, false, b1,
                                                   (short)0, acc01, false, false);
    acc10 = __builtin_amdgcn_wmma_f32_16x16x32_f16(false, a1, false, b0,
                                                   (short)0, acc10, false, false);
    acc11 = __builtin_amdgcn_wmma_f32_16x16x32_f16(false, a1, false, b1,
                                                   (short)0, acc11, false, false);
  }

  const float aa = a_sc[0];
  const int col0 = tn * 32 + (lane & 15);
  const float cs0 = 0.5f * aa * aa * rsqrtf(k2[col0]);
  const float cs1 = 0.5f * aa * aa * rsqrtf(k2[col0 + 16]);
  const int rowOff = (lane < 16) ? 0 : 8;
#pragma unroll
  for (int r = 0; r < 8; ++r) {
    int row0 = tm * 32 + rowOff + r;
    float rs0 = rsqrtf(k1[row0]);
    float rs1 = rsqrtf(k1[row0 + 16]);
    Kout[(size_t)row0 * NS + col0]             = acc00[r] * cs0 * rs0;
    Kout[(size_t)row0 * NS + col0 + 16]        = acc01[r] * cs1 * rs0;
    Kout[(size_t)(row0 + 16) * NS + col0]      = acc10[r] * cs0 * rs1;
    Kout[(size_t)(row0 + 16) * NS + col0 + 16] = acc11[r] * cs1 * rs1;
  }
}

// ---------------------------------------------------------------------------
extern "C" void kernel_launch(void* const* d_in, const int* in_sizes, int n_in,
                              void* d_out, int out_size, void* d_ws, size_t ws_size,
                              hipStream_t stream) {
  const int*   X1 = (const int*)d_in[0];
  const int*   X2 = (const int*)d_in[1];
  const float* E  = (const float*)d_in[2];
  const float* W  = (const float*)d_in[3];
  const float* a  = (const float*)d_in[4];
  float* Kout = (float*)d_out;

  // Workspace carve-up (all offsets 256B-aligned), ~39.7 MB total.
  char* ws = (char*)d_ws;
  _Float16* Wt  = (_Float16*)(ws);                         size_t off = (size_t)ND * KDIM * 2;   // 16,384,000
  float*    V1  = (float*)(ws + off);                      off += (size_t)NS * ND * 4;           //  2,621,440
  float*    V2  = (float*)(ws + off);                      off += (size_t)NS * ND * 4;
  float*    S1  = (float*)(ws + off);                      off += (size_t)NS * N_AA * N_AA * 4;  //    819,200
  float*    S2  = (float*)(ws + off);                      off += (size_t)NS * N_AA * N_AA * 4;
  float*    k1  = (float*)(ws + off);                      off += 2048;
  float*    k2  = (float*)(ws + off);                      off += 2048;
  _Float16* Ab  = (_Float16*)(ws + off);                   off += (size_t)NS * KBIG * 2;         //  8,192,000
  _Float16* Bb  = (_Float16*)(ws + off);                   off += (size_t)NS * KBIG * 2;
  (void)ws_size; (void)in_sizes; (void)n_in; (void)out_size;

  // 0) W -> Wt (f16, transposed)
  transpose_w<<<(ND / 32) * (KDIM / 32), 256, 0, stream>>>(W, Wt);

  // 1) V1, V2 via fused-gather WMMA GEMM  (640 waves = 80 blocks each)
  net_gemm<<<(NS / 16) * (ND / 64) / 8, 256, 0, stream>>>(X1, E, Wt, V1);
  net_gemm<<<(NS / 16) * (ND / 64) / 8, 256, 0, stream>>>(X2, E, Wt, V2);

  // 2) Gram matrices
  gram<<<NS, 256, 0, stream>>>(V1, S1);
  gram<<<NS, 256, 0, stream>>>(V2, S2);

  // 3) diagonal sums
  kdiag<<<(NS + 255) / 256, 256, 0, stream>>>(X1, S1, k1, NS);
  kdiag<<<(NS + 255) / 256, 256, 0, stream>>>(X2, S2, k2, NS);

  // 4) build cross-kernel operands
  {
    size_t tot = (size_t)NS * LSEQ * N_AA;  // 2,048,000
    int blocks = (int)((tot + 255) / 256);
    build_A<<<blocks, 256, 0, stream>>>(X1, S1, Ab);
    build_B<<<blocks, 256, 0, stream>>>(X2, S2, Bb);
  }

  // 5) final 512x512x8000 WMMA GEMM (32x32 register tiles) + fused normalization
  cross_gemm<<<(NS / 32) * (NS / 32) / 8, 256, 0, stream>>>(Ab, Bb, k1, k2, a, Kout);
}
